// Pips_28355374088987
// MI455X (gfx1250) — matrix operand order repaired and checked by reading, not compiled, not run on hardware
//
#include <hip/hip_runtime.h>
#include <hip/hip_bf16.h>
#include <math.h>

// ---------------------------------------------------------------------------
// PIPs forward for MI455X (gfx1250, wave32, WMMA).
// Heavy GEMMs -> v_wmma_f32_16x16x32_bf16 with pre-packed (fragment-order)
// B operands: every lane's B fragment is 32 contiguous bytes (2x b128 loads).
// Each wave computes a 32x64 output block (2 M-tiles x 4 N-tiles, 8 accs).
// ---------------------------------------------------------------------------

typedef __bf16 bf16_t;
typedef __attribute__((ext_vector_type(16))) __bf16 v16bf;
typedef __attribute__((ext_vector_type(8)))  float  v8f;

__device__ __forceinline__ float geluf(float x) {
    return 0.5f * x * (1.0f + erff(x * 0.70710678118654752f));
}

// ---------------------------------------------------------------------------
// Packed-B WMMA GEMM: C[M,N] = epilogue(scale * A[M,K] @ B[K,N]).
// A: bf16 row-major (lda). Bp: bf16 packed fragment order:
//   Bp[((kt*(N/16) + nt) * 512) + lane*16 + j] = B[kt*32 + (lane>>4)*16 + j][nt*16 + (lane&15)]
// C: f32 row-major (ldc); optional bf16 mirror Cb.
// M % 32 == 0, N % 64 == 0, K % 32 == 0.
// EPI: 0 = scale only; 1 = +bias[col]; 2 = gelu(+bias); 3 = +bias + res[off].
// ---------------------------------------------------------------------------
template <int EPI>
__global__ __launch_bounds__(256) void gemm_wmma_pk(
    const bf16_t* A, const bf16_t* Bp, float* C, bf16_t* Cb,
    const float* bias, const float* res,
    int M, int N, int K, int lda, int ldc,
    long sA, long sB, long sC, float scale)
{
    int batch = blockIdx.y;
    A  += (long)batch * sA;
    Bp += (long)batch * sB;
    long cOff = (long)batch * sC;

    int tilesN = N >> 4;
    int n4     = N >> 6;
    int wtiles = (M >> 5) * n4;
    int wave   = threadIdx.x >> 5;
    int wt     = blockIdx.x * 8 + wave;
    if (wt >= wtiles) return;            // uniform per wave: EXEC stays full
    int tm  = (wt / n4) * 2;             // two 16-row tiles
    int tn0 = (wt - (wt / n4) * n4) * 4; // four 16-col tiles

    int lane = threadIdx.x & 31;
    int half = lane >> 4;                // 0: lanes 0-15, 1: lanes 16-31
    int l15  = lane & 15;

    // A fragment (16x32): row tm*16+l15; VGPR0-7 K=half*16+{0..15}
    const bf16_t* arow0 = A + (long)(tm * 16 + l15) * lda + half * 16;
    const bf16_t* arow1 = arow0 + (long)16 * lda;
    const bf16_t* bbase = Bp + ((long)lane << 4);

    union U32x8 { uint32_t u[8]; v16bf v; };
    v8f acc[8] = {};                     // acc[mi*4 + t]

    for (int k0 = 0; k0 < K; k0 += 32) {
        U32x8 a0, a1;
        {
            const uint32_t* p = (const uint32_t*)(arow0 + k0);
            const uint32_t* q = (const uint32_t*)(arow0 + k0 + 8);
            a0.u[0] = p[0]; a0.u[1] = p[1]; a0.u[2] = p[2]; a0.u[3] = p[3];
            a0.u[4] = q[0]; a0.u[5] = q[1]; a0.u[6] = q[2]; a0.u[7] = q[3];
        }
        {
            const uint32_t* p = (const uint32_t*)(arow1 + k0);
            const uint32_t* q = (const uint32_t*)(arow1 + k0 + 8);
            a1.u[0] = p[0]; a1.u[1] = p[1]; a1.u[2] = p[2]; a1.u[3] = p[3];
            a1.u[4] = q[0]; a1.u[5] = q[1]; a1.u[6] = q[2]; a1.u[7] = q[3];
        }
        long tb = (((long)(k0 >> 5)) * tilesN + tn0) << 9;
        __builtin_prefetch(bbase + tb + ((long)tilesN << 9), 0, 1);
        U32x8 b[4];
#pragma unroll
        for (int t = 0; t < 4; ++t) {
            const uint32_t* p = (const uint32_t*)(bbase + tb + ((long)t << 9));
#pragma unroll
            for (int j = 0; j < 8; ++j) b[t].u[j] = p[j];
        }
#pragma unroll
        for (int t = 0; t < 4; ++t) {
            acc[t]     = __builtin_amdgcn_wmma_f32_16x16x32_bf16(
                false, a0.v, false, b[t].v, (short)0, acc[t],     false, false);
            acc[4 + t] = __builtin_amdgcn_wmma_f32_16x16x32_bf16(
                false, a1.v, false, b[t].v, (short)0, acc[4 + t], false, false);
        }
    }

    // D layout: VGPR r -> row half*8 + r, col = l15
#pragma unroll
    for (int mi = 0; mi < 2; ++mi) {
#pragma unroll
        for (int t = 0; t < 4; ++t) {
            int col = (tn0 + t) * 16 + l15;
#pragma unroll
            for (int r = 0; r < 8; ++r) {
                int row = (tm + mi) * 16 + half * 8 + r;
                long o = (long)row * ldc + col;
                float v = acc[mi * 4 + t][r] * scale;
                if (EPI >= 1) v += bias[col];
                if (EPI == 2) v = geluf(v);
                if (EPI == 3) v += res[cOff + o];
                if (C)  C[cOff + o]  = v;
                if (Cb) Cb[cOff + o] = (bf16_t)v;
            }
        }
    }
}

// ---------------------------------------------------------------------------
// Pack f32 B[K][N] (per batch, zero-padded to K x N from Ksrc x Nsrc) into
// bf16 WMMA fragment order (see gemm_wmma_pk header comment).
// ---------------------------------------------------------------------------
__global__ void k_pack_w(const float* W, bf16_t* out, int K, int N,
                         int Ksrc, int Nsrc, int nb) {
    long i = (long)blockIdx.x * blockDim.x + threadIdx.x;
    long per = (long)K * N;
    if (i >= per * nb) return;
    int b = (int)(i / per);
    long rkn = i - (long)b * per;
    int k = (int)(rkn / N);
    int n = (int)(rkn - (long)k * N);
    float v = (k < Ksrc && n < Nsrc) ? W[(long)b * Ksrc * Nsrc + (long)k * Nsrc + n]
                                     : 0.0f;
    int kt = k >> 5, kin = k & 31, hf = kin >> 4, j = kin & 15;
    int nt = n >> 4, col = n & 15;
    long dst = (long)b * per
             + (((long)kt * (N >> 4) + nt) << 9) + (long)((hf * 16 + col) << 4) + j;
    out[dst] = (bf16_t)v;
}

// pad out_b (1040) -> f32 (1088), zero tail
__global__ void k_pad_outb(const float* b, float* out) {
    int i = blockIdx.x * blockDim.x + threadIdx.x;
    if (i >= 1088) return;
    out[i] = (i < 1040) ? b[i] : 0.0f;
}

// ---------------------------------------------------------------------------
// fnet: 7x7 stride-4 SAME conv + ReLU.  rgb [8][3][384][512] -> fm [8][128][96][128]
// ---------------------------------------------------------------------------
__global__ void k_fnet(const float* rgb, const float* w, const float* b, float* fm) {
    int idx = blockIdx.x * blockDim.x + threadIdx.x;
    if (idx >= 8 * 128 * 96 * 128) return;
    int x = idx & 127; int t = idx >> 7;
    int y = t % 96; t /= 96;
    int c = t & 127; int s = t >> 7;
    float acc = b[c];
    const float* wp = w + c * 3 * 49;
    for (int ic = 0; ic < 3; ++ic) {
        const float* ip = rgb + (long)(s * 3 + ic) * 384 * 512;
        for (int ky = 0; ky < 7; ++ky) {
            int iy = y * 4 - 1 + ky;
            if (iy < 0 || iy >= 384) continue;
            for (int kx = 0; kx < 7; ++kx) {
                int ix = x * 4 - 1 + kx;
                if (ix < 0 || ix >= 512) continue;
                acc += ip[iy * 512 + ix] * wp[ic * 49 + ky * 7 + kx];
            }
        }
    }
    fm[idx] = fmaxf(acc, 0.0f);
}

__global__ void k_coords_init(const float* xys, float* coords) {
    int idx = blockIdx.x * blockDim.x + threadIdx.x;
    if (idx >= 4096) return;
    int k = idx & 1, n = (idx >> 1) & 255;
    coords[idx] = xys[n * 2 + k] * 0.25f;
}

// ff0: bilinear (clamp, no zero-pad) sample of fmaps[s=0]; broadcast over s.
__global__ void k_ff0(const float* xys, const float* fm, float* ffe, bf16_t* ffeb) {
    int idx = blockIdx.x * blockDim.x + threadIdx.x;
    if (idx >= 256 * 128) return;
    int c = idx & 127, n = idx >> 7;
    float X = xys[n * 2 + 0] * 0.25f, Y = xys[n * 2 + 1] * 0.25f;
    float x0 = floorf(X), y0 = floorf(Y);
    float wx = X - x0, wy = Y - y0;
    const float* img = fm + (long)c * 96 * 128;   // s = 0
    auto g = [&](float xi, float yi) -> float {
        int xc = (int)fminf(fmaxf(xi, 0.0f), 127.0f);
        int yc = (int)fminf(fmaxf(yi, 0.0f), 95.0f);
        return img[yc * 128 + xc];
    };
    float v = g(x0, y0) * (1 - wx) * (1 - wy) + g(x0 + 1, y0) * wx * (1 - wy)
            + g(x0, y0 + 1) * (1 - wx) * wy + g(x0 + 1, y0 + 1) * wx * wy;
    for (int s = 0; s < 8; ++s) {
        long o = ((long)(s * 256 + n)) * 128 + c;
        ffe[o] = v;
        ffeb[o] = (bf16_t)v;
    }
}

// 2x2 average pool on [2048][Hi][Wi]
__global__ void k_pool2(const float* in, float* out, int Hi, int Wi, long total) {
    long i = (long)blockIdx.x * blockDim.x + threadIdx.x;
    if (i >= total) return;
    int Wo = Wi >> 1, Ho = Hi >> 1;
    int x = (int)(i % Wo); long t = i / Wo;
    int y = (int)(t % Ho); long sn = t / Ho;
    const float* p = in + (sn * Hi + 2 * y) * Wi + 2 * x;
    out[i] = 0.25f * (p[0] + p[1] + p[Wi] + p[Wi + 1]);
}

// ffeats -> xin cols [0,128)
__global__ void k_xin_ff(const float* ffe, bf16_t* xin) {
    int idx = blockIdx.x * blockDim.x + threadIdx.x;
    if (idx >= 2048 * 128) return;
    int c = idx & 127, t = idx >> 7, s = t & 7, n = t >> 3;
    xin[(long)(n * 8 + s) * 544 + c] = (bf16_t)ffe[((long)(s * 256 + n)) * 128 + c];
}

// fcorr sampling (zero-pad bilinear) -> xin cols [128,324)
__global__ void k_fcorr(const float* coords, const float* c0, const float* c1,
                        const float* c2, const float* c3, bf16_t* xin) {
    int idx = blockIdx.x * blockDim.x + threadIdx.x;
    if (idx >= 8 * 256 * 4 * 49) return;
    int p = idx % 49; int t = idx / 49;
    int l = t & 3; t >>= 2;
    int n = t & 255; int s = t >> 8;
    int dy = p / 7 - 3, dx = p % 7 - 3;
    const float* cv; int Hi, Wi;
    if (l == 0)      { cv = c0; Hi = 96; Wi = 128; }
    else if (l == 1) { cv = c1; Hi = 48; Wi = 64; }
    else if (l == 2) { cv = c2; Hi = 24; Wi = 32; }
    else             { cv = c3; Hi = 12; Wi = 16; }
    float inv = 1.0f / (float)(1 << l);
    float X = coords[(s * 256 + n) * 2 + 0] * inv + (float)dx;
    float Y = coords[(s * 256 + n) * 2 + 1] * inv + (float)dy;
    const float* img = cv + ((long)(s * 256 + n)) * Hi * Wi;
    float x0 = floorf(X), y0 = floorf(Y), wx = X - x0, wy = Y - y0;
    auto g = [&](float xi, float yi) -> float {
        int xc = (int)fminf(fmaxf(xi, 0.0f), (float)(Wi - 1));
        int yc = (int)fminf(fmaxf(yi, 0.0f), (float)(Hi - 1));
        float v = img[yc * Wi + xc];
        bool valid = (xi >= 0.0f) && (xi <= (float)(Wi - 1)) &&
                     (yi >= 0.0f) && (yi <= (float)(Hi - 1));
        return valid ? v : 0.0f;
    };
    float v = g(x0, y0) * (1 - wx) * (1 - wy) + g(x0 + 1, y0) * wx * (1 - wy)
            + g(x0, y0 + 1) * (1 - wx) * wy + g(x0 + 1, y0 + 1) * wx * wy;
    xin[(long)(n * 8 + s) * 544 + 128 + l * 49 + p] = (bf16_t)v;
}

// sin/cos positional embedding -> xin cols [324,519), zero pad [519,544)
__global__ void k_xin_emb(const float* coords, bf16_t* xin) {
    int idx = blockIdx.x * blockDim.x + threadIdx.x;
    if (idx >= 2048) return;
    int s = idx & 7, n = idx >> 3;
    float fx = coords[(s * 256 + n) * 2 + 0] - coords[n * 2 + 0];
    float fy = coords[(s * 256 + n) * 2 + 1] - coords[n * 2 + 1];
    float tt = 8.0f * (float)s / 7.0f;               // linspace(0, 8, 8)
    bf16_t* o = xin + (long)(n * 8 + s) * 544 + 324;
    float v3[3] = {fx, fy, tt};
    for (int cc = 0; cc < 3; ++cc) {
        for (int j = 0; j < 32; ++j) {
            float a = v3[cc] * (31.25f * (float)j);  // arange(0,64,2)*1000/64
            o[cc * 64 + 2 * j]     = (bf16_t)sinf(a);
            o[cc * 64 + 2 * j + 1] = (bf16_t)cosf(a);
        }
    }
    o[192] = (bf16_t)fx; o[193] = (bf16_t)fy; o[194] = (bf16_t)tt;
    for (int j = 195; j < 220; ++j) o[j] = (bf16_t)0.0f;   // K pad 519..543
}

// layernorm over 512 per token (wave per token); writes f32 + bf16
__global__ __launch_bounds__(256) void k_layernorm(
    const float* x, const float* g, const float* b,
    float* h, bf16_t* hb, int ntok) {
    int wave = threadIdx.x >> 5, lane = threadIdx.x & 31;
    int tok = blockIdx.x * 8 + wave;
    if (tok >= ntok) return;
    const float* xp = x + (long)tok * 512;
    float v[16], sum = 0.0f, sum2 = 0.0f;
#pragma unroll
    for (int j = 0; j < 16; ++j) {
        float t = xp[lane + 32 * j];
        v[j] = t; sum += t; sum2 += t * t;
    }
    for (int o = 16; o > 0; o >>= 1) {
        sum  += __shfl_xor(sum, o, 32);
        sum2 += __shfl_xor(sum2, o, 32);
    }
    float m  = sum * (1.0f / 512.0f);
    float var = sum2 * (1.0f / 512.0f) - m * m;
    float rs = rsqrtf(var + 1e-5f);
#pragma unroll
    for (int j = 0; j < 16; ++j) {
        int c = lane + 32 * j;
        float t = (v[j] - m) * rs * g[c] + b[c];
        h[(long)tok * 512 + c]  = t;
        hb[(long)tok * 512 + c] = (bf16_t)t;
    }
}

// token mixing: x += (gelu(h @s tw1 + tb1)) @t tw2 + tb2   (K=8, VALU)
__global__ void k_tokenmix(const float* h, const float* tw1, const float* tb1,
                           const float* tw2, const float* tb2, float* x) {
    int idx = blockIdx.x * blockDim.x + threadIdx.x;
    if (idx >= 256 * 512) return;
    int d = idx & 511, m = idx >> 9;
    float hv[8];
#pragma unroll
    for (int s = 0; s < 8; ++s) hv[s] = h[((long)(m * 8 + s)) * 512 + d];
    float outs[8];
#pragma unroll
    for (int s = 0; s < 8; ++s) outs[s] = tb2[s];
    for (int t = 0; t < 32; ++t) {
        float u = tb1[t];
#pragma unroll
        for (int s = 0; s < 8; ++s) u += hv[s] * tw1[s * 32 + t];
        u = geluf(u);
#pragma unroll
        for (int s = 0; s < 8; ++s) outs[s] += u * tw2[t * 8 + s];
    }
#pragma unroll
    for (int s = 0; s < 8; ++s) x[((long)(m * 8 + s)) * 512 + d] += outs[s];
}

// mean over S of h -> xmean bf16 [256][512]
__global__ void k_meanS(const float* h, bf16_t* xmb) {
    int idx = blockIdx.x * blockDim.x + threadIdx.x;
    if (idx >= 256 * 512) return;
    int c = idx & 511, m = idx >> 9;
    float a = 0.0f;
#pragma unroll
    for (int s = 0; s < 8; ++s) a += h[((long)(m * 8 + s)) * 512 + c];
    xmb[(long)m * 512 + c] = (bf16_t)(a * 0.125f);
}

// coords += delta[...,:2]; write coords*4 to cp and cp2 slots (delta ld 1088)
__global__ void k_coords_update(float* coords, const float* delta,
                                float* cp_dst, float* cp2_dst) {
    int idx = blockIdx.x * blockDim.x + threadIdx.x;
    if (idx >= 4096) return;
    int k = idx & 1, n = (idx >> 1) & 255, s = idx >> 9;
    float c = coords[idx] + delta[(long)n * 1088 + s * 130 + k];
    coords[idx] = c;
    cp_dst[idx]  = c * 4.0f;
    cp2_dst[idx] = c * 4.0f;
}

__global__ void k_cp2_init(const float* xys, float* dst) {
    int idx = blockIdx.x * blockDim.x + threadIdx.x;
    if (idx >= 4096) return;
    int k = idx & 1, n = (idx >> 1) & 255;
    float v = xys[n * 2 + k];
    dst[idx] = v; dst[4096 + idx] = v;
}

__global__ void k_cp2_final(const float* coords, float* dst) {
    int idx = blockIdx.x * blockDim.x + threadIdx.x;
    if (idx >= 4096) return;
    float v = coords[idx] * 4.0f;
    dst[idx] = v; dst[4096 + idx] = v;
}

// groupnorm(1,128) over delta[...,2:] -> bf16 g  (wave per row, row = n*8+s)
__global__ __launch_bounds__(256) void k_groupnorm(
    const float* delta, const float* gg, const float* gb, bf16_t* g) {
    int wave = threadIdx.x >> 5, lane = threadIdx.x & 31;
    int row = blockIdx.x * 8 + wave;
    if (row >= 2048) return;
    int n = row >> 3, s = row & 7;
    const float* dp = delta + (long)n * 1088 + s * 130 + 2;
    float v[4], sum = 0.0f, sum2 = 0.0f;
#pragma unroll
    for (int j = 0; j < 4; ++j) {
        float t = dp[lane + 32 * j];
        v[j] = t; sum += t; sum2 += t * t;
    }
    for (int o = 16; o > 0; o >>= 1) {
        sum  += __shfl_xor(sum, o, 32);
        sum2 += __shfl_xor(sum2, o, 32);
    }
    float m = sum * (1.0f / 128.0f);
    float var = sum2 * (1.0f / 128.0f) - m * m;
    float rs = rsqrtf(var + 1e-5f);
#pragma unroll
    for (int j = 0; j < 4; ++j) {
        int c = lane + 32 * j;
        g[(long)row * 128 + c] = (bf16_t)((v[j] - m) * rs * gg[c] + gb[c]);
    }
}

// ffeats[s][n] += updo[n*8+s]  (residual), refresh bf16 copy
__global__ void k_ffeat_update(const float* updo, float* ffe, bf16_t* ffeb) {
    int idx = blockIdx.x * blockDim.x + threadIdx.x;
    if (idx >= 2048 * 128) return;
    int c = idx & 127, t = idx >> 7, s = t & 7, n = t >> 3;
    long o = ((long)(s * 256 + n)) * 128 + c;
    float v = ffe[o] + updo[((long)(n * 8 + s)) * 128 + c];
    ffe[o] = v;
    ffeb[o] = (bf16_t)v;
}

// vis[s*256+n] = ffeats[s][n] . vis_w + vis_b   (wave per row)
__global__ __launch_bounds__(256) void k_vis(const float* ffe, const float* vw,
                                             const float* vb, float* out) {
    int wave = threadIdx.x >> 5, lane = threadIdx.x & 31;
    int row = blockIdx.x * 8 + wave;
    if (row >= 2048) return;
    float a = 0.0f;
#pragma unroll
    for (int j = 0; j < 4; ++j) {
        int c = lane + 32 * j;
        a += ffe[(long)row * 128 + c] * vw[c];
    }
    for (int o = 16; o > 0; o >>= 1) a += __shfl_xor(a, o, 32);
    if (lane == 0) out[row] = a + vb[0];
}

// ---------------------------------------------------------------------------
// Host orchestration
// ---------------------------------------------------------------------------
extern "C" void kernel_launch(void* const* d_in, const int* in_sizes, int n_in,
                              void* d_out, int out_size, void* d_ws, size_t ws_size,
                              hipStream_t stream) {
    (void)in_sizes; (void)n_in; (void)out_size; (void)ws_size;

    const float* xys  = (const float*)d_in[0];
    const float* rgbs = (const float*)d_in[1];
    int pi = 2;
    const float* fnet_w = (const float*)d_in[pi++];
    const float* fnet_b = (const float*)d_in[pi++];
    const float* in_w   = (const float*)d_in[pi++];
    const float* in_b   = (const float*)d_in[pi++];
    const float* ln1_g  = (const float*)d_in[pi++];
    const float* ln1_b  = (const float*)d_in[pi++];
    const float* tw1    = (const float*)d_in[pi++];
    const float* tb1    = (const float*)d_in[pi++];
    const float* tw2    = (const float*)d_in[pi++];
    const float* tb2    = (const float*)d_in[pi++];
    const float* ln2_g  = (const float*)d_in[pi++];
    const float* ln2_b  = (const float*)d_in[pi++];
    const float* cw1    = (const float*)d_in[pi++];
    const float* cb1    = (const float*)d_in[pi++];
    const float* cw2    = (const float*)d_in[pi++];
    const float* cb2    = (const float*)d_in[pi++];
    const float* oln_g  = (const float*)d_in[pi++];
    const float* oln_b  = (const float*)d_in[pi++];
    const float* out_w  = (const float*)d_in[pi++];
    const float* out_b  = (const float*)d_in[pi++];
    const float* gn_g   = (const float*)d_in[pi++];
    const float* gn_b   = (const float*)d_in[pi++];
    const float* upd_w  = (const float*)d_in[pi++];
    const float* upd_b  = (const float*)d_in[pi++];
    const float* vis_w  = (const float*)d_in[pi++];
    const float* vis_b  = (const float*)d_in[pi++];
    float* out = (float*)d_out;

    // workspace carve-up
    char* wsb = (char*)d_ws;
    size_t off = 0;
    auto wsalloc = [&](size_t bytes) -> void* {
        void* p = (void*)(wsb + off);
        off += (bytes + 255) & ~(size_t)255;
        return p;
    };
    float*  fm    = (float*) wsalloc(12582912ull * 4);   // [8][128][96][128]
    bf16_t* fmb   = (bf16_t*)wsalloc(12582912ull * 2);   // packed per s
    float*  ffe   = (float*) wsalloc(262144ull * 4);     // [8][256][128]
    bf16_t* ffeb  = (bf16_t*)wsalloc(262144ull * 2);
    float*  coords= (float*) wsalloc(4096ull * 4);       // [8][256][2]
    float*  c0    = (float*) wsalloc(25165824ull * 4);   // [2048][96][128]
    float*  c1    = (float*) wsalloc(6291456ull * 4);
    float*  c2    = (float*) wsalloc(1572864ull * 4);
    float*  c3    = (float*) wsalloc(393216ull * 4);
    bf16_t* xin   = (bf16_t*)wsalloc(2048ull * 544 * 2);
    float*  x     = (float*) wsalloc(2048ull * 512 * 4);
    float*  h     = (float*) wsalloc(2048ull * 512 * 4);
    bf16_t* hb    = (bf16_t*)wsalloc(2048ull * 512 * 2);
    bf16_t* ub    = (bf16_t*)wsalloc(2048ull * 2048 * 2);
    bf16_t* xmb   = (bf16_t*)wsalloc(256ull * 512 * 2);
    float*  delta = (float*) wsalloc(256ull * 1088 * 4); // ld 1088 (padded N)
    bf16_t* gbuf  = (bf16_t*)wsalloc(2048ull * 128 * 2);
    float*  updo  = (float*) wsalloc(2048ull * 128 * 4);
    bf16_t* inwb  = (bf16_t*)wsalloc(544ull * 512 * 2);      // packed
    bf16_t* cw1b  = (bf16_t*)wsalloc(12ull * 512 * 2048 * 2);
    bf16_t* cw2b  = (bf16_t*)wsalloc(12ull * 2048 * 512 * 2);
    bf16_t* outwb = (bf16_t*)wsalloc(512ull * 1088 * 2);
    bf16_t* updwb = (bf16_t*)wsalloc(128ull * 128 * 2);
    float*  outbp = (float*) wsalloc(1088ull * 4);

    auto pack = [&](const float* src, bf16_t* dst, int K, int N, int Ks, int Ns, int nb) {
        long total = (long)K * N * nb;
        k_pack_w<<<(int)((total + 255) / 256), 256, 0, stream>>>(src, dst, K, N, Ks, Ns, nb);
    };
    auto gemm = [&](int epi, const bf16_t* A, const bf16_t* B, float* C, bf16_t* Cb,
                    const float* bias, const float* res,
                    int M, int N, int K, int lda, int ldc,
                    long sA, long sB, long sC, int batches, float scale) {
        int wtiles = (M >> 5) * (N >> 6);
        dim3 grid((unsigned)((wtiles + 7) / 8), (unsigned)batches);
        switch (epi) {
        case 0: gemm_wmma_pk<0><<<grid, 256, 0, stream>>>(A, B, C, Cb, bias, res, M, N, K, lda, ldc, sA, sB, sC, scale); break;
        case 1: gemm_wmma_pk<1><<<grid, 256, 0, stream>>>(A, B, C, Cb, bias, res, M, N, K, lda, ldc, sA, sB, sC, scale); break;
        case 2: gemm_wmma_pk<2><<<grid, 256, 0, stream>>>(A, B, C, Cb, bias, res, M, N, K, lda, ldc, sA, sB, sC, scale); break;
        default: gemm_wmma_pk<3><<<grid, 256, 0, stream>>>(A, B, C, Cb, bias, res, M, N, K, lda, ldc, sA, sB, sC, scale); break;
        }
    };

    // ---- weight packing (bf16 fragment order; L2-resident)
    pack(in_w,  inwb,  544, 512,  519, 512,  1);
    pack(cw1,   cw1b,  512, 2048, 512, 2048, 12);
    pack(cw2,   cw2b,  2048, 512, 2048, 512, 12);
    pack(out_w, outwb, 512, 1088, 512, 1040, 1);
    pack(upd_w, updwb, 128, 128,  128, 128,  1);
    k_pad_outb<<<5, 256, 0, stream>>>(out_b, outbp);

    // ---- encoder + init
    k_fnet<<<12582912 / 256, 256, 0, stream>>>(rgbs, fnet_w, fnet_b, fm);
    pack(fm, fmb, 128, 12288, 128, 12288, 8);     // fmaps as packed B, per s
    k_coords_init<<<16, 256, 0, stream>>>(xys, coords);
    k_ff0<<<128, 256, 0, stream>>>(xys, fm, ffe, ffeb);
    k_cp2_init<<<16, 256, 0, stream>>>(xys, out + 12288);   // cp2[0], cp2[1]

    const float inv_sqrtC = 0.08838834764831845f;            // 1/sqrt(128)

    for (int iter = 0; iter < 3; ++iter) {
        // correlation volume per frame s: [256,12288] = ffeats[s] @ fmaps[s]
        gemm(0, ffeb, fmb, c0, nullptr, nullptr, nullptr,
             256, 12288, 128, 128, 12288,
             256L * 128, 128L * 12288, 256L * 12288, 8, inv_sqrtC);
        // pyramid
        k_pool2<<<(int)(2048L * 48 * 64 / 256), 256, 0, stream>>>(c0, c1, 96, 128, 2048L * 48 * 64);
        k_pool2<<<(int)(2048L * 24 * 32 / 256), 256, 0, stream>>>(c1, c2, 48, 64, 2048L * 24 * 32);
        k_pool2<<<(int)(2048L * 12 * 16 / 256), 256, 0, stream>>>(c2, c3, 24, 32, 2048L * 12 * 16);
        // build xin [2048][544] bf16
        k_xin_ff<<<1024, 256, 0, stream>>>(ffe, xin);
        k_fcorr<<<1568, 256, 0, stream>>>(coords, c0, c1, c2, c3, xin);
        k_xin_emb<<<8, 256, 0, stream>>>(coords, xin);
        // mixer in-proj: x = xin @ in_w + in_b
        gemm(1, xin, inwb, x, nullptr, in_b, nullptr,
             2048, 512, 544, 544, 512, 0, 0, 0, 1, 1.0f);
        for (int d = 0; d < 12; ++d) {
            k_layernorm<<<256, 256, 0, stream>>>(x, ln1_g + d * 512, ln1_b + d * 512, h, hb, 2048);
            k_tokenmix<<<512, 256, 0, stream>>>(h, tw1 + d * 256, tb1 + d * 32,
                                                tw2 + d * 256, tb2 + d * 8, x);
            k_layernorm<<<256, 256, 0, stream>>>(x, ln2_g + d * 512, ln2_b + d * 512, h, hb, 2048);
            // u = gelu(h @ cw1 + cb1)  (bf16 out, fused)
            gemm(2, hb, cw1b + (long)d * 512 * 2048, nullptr, ub, cb1 + d * 2048, nullptr,
                 2048, 2048, 512, 512, 2048, 0, 0, 0, 1, 1.0f);
            // x = x + u @ cw2 + cb2
            gemm(3, ub, cw2b + (long)d * 2048 * 512, x, nullptr, cb2 + d * 512, x,
                 2048, 512, 2048, 2048, 512, 0, 0, 0, 1, 1.0f);
        }
        k_layernorm<<<256, 256, 0, stream>>>(x, oln_g, oln_b, h, hb, 2048);
        k_meanS<<<512, 256, 0, stream>>>(h, xmb);
        // delta = xmean @ out_w + out_b   [256,1088] (cols >=1040 are pad)
        gemm(1, xmb, outwb, delta, nullptr, outbp, nullptr,
             256, 1088, 512, 512, 1088, 0, 0, 0, 1, 1.0f);
        // coords update; write cp[iter], cp2[2+iter]
        k_coords_update<<<16, 256, 0, stream>>>(coords, delta,
                                                out + (long)iter * 4096,
                                                out + 12288 + (long)(2 + iter) * 4096);
        // feature update: ffeats += gelu(groupnorm(delta[...,2:]) @ upd_w + upd_b)
        k_groupnorm<<<256, 256, 0, stream>>>(delta, gn_g, gn_b, gbuf);
        gemm(2, gbuf, updwb, updo, nullptr, upd_b, nullptr,
             2048, 128, 128, 128, 128, 0, 0, 0, 1, 1.0f);
        k_ffeat_update<<<1024, 256, 0, stream>>>(updo, ffe, ffeb);
    }
    k_cp2_final<<<16, 256, 0, stream>>>(coords, out + 12288 + 5L * 4096);  // cp2[5], cp2[6]
    k_vis<<<256, 256, 0, stream>>>(ffe, vis_w, vis_b, out + 40960);
}
